// TransformerLayer_17678085391074
// MI455X (gfx1250) — matrix operand order repaired and checked by reading, not compiled, not run on hardware
//
#include <hip/hip_runtime.h>
#include <math.h>
#include <stdint.h>

// ---------------- problem constants ----------------
constexpr int Bm  = 2;
constexpr int Tm  = 2048;
constexpr int Dm  = 1024;
constexpr int Hm  = 16;
constexpr int HDm = 64;           // head dim
constexpr int Mm  = Bm * Tm;      // 4096 rows
constexpr int Fm  = 4 * Dm;       // 4096 ffn dim
constexpr float EPSV = 1e-5f;

typedef __bf16 bf16;
typedef __attribute__((ext_vector_type(16))) __bf16 v16bf;
typedef __attribute__((ext_vector_type(8)))  float  v8f;

static __device__ __forceinline__ v8f wmma_bf16(v16bf a, v16bf b, v8f c) {
  return __builtin_amdgcn_wmma_f32_16x16x32_bf16(
      false, a, false, b, (short)0, c, false, false);
}

// A-matrix (16x32, 16-bit) K pattern per ISA 7.12.2.
static __device__ __forceinline__ int kpat(int d) {
  return (d < 4) ? (2 * d) : (16 + 2 * (d - 4));
}

// Load A-fragment (rows rowStart..+15, k kbase..+31) from LDS [row][k] layout.
static __device__ __forceinline__ v16bf lds_Afrag(const bf16* base, int rowStart,
                                                  int rowStride, int kbase, int lane) {
  const int m = lane & 15;
  const int halfo = (lane >> 4) << 3;   // lanes 16-31 hold K+8
  union { v16bf v; unsigned int u[8]; } r;
  const bf16* p = base + (size_t)(rowStart + m) * rowStride + kbase;
#pragma unroll
  for (int d = 0; d < 8; ++d)
    r.u[d] = *(const unsigned int*)(p + kpat(d) + halfo);
  return r.v;
}

// Load B-fragment (n = nbase + lane%16, k in A table order) from LDS [k][n] layout.
static __device__ __forceinline__ v16bf lds_Bfrag(const bf16* base, int rowStride,
                                                  int nbase, int lane) {
  union { v16bf v; bf16 h[16]; } r;
  const int halfo = (lane >> 4) << 3;
  const bf16* p = base + nbase + (lane & 15);
#pragma unroll
  for (int s = 0; s < 16; ++s)
    r.h[s] = p[(size_t)(((s < 8) ? s : (s + 8)) + halfo) * rowStride];
  return r.v;
}

// Per-lane 16B async copy global -> LDS (ASYNCcnt-tracked).
static __device__ __forceinline__ void async_copy_b128(bf16* lds_dst, const bf16* gsrc) {
  unsigned ldst = (unsigned)(uintptr_t)lds_dst;
  asm volatile("global_load_async_to_lds_b128 %0, %1, off"
               :: "v"(ldst), "v"(gsrc) : "memory");
}
// ASYNC loads retire in order: waiting <=N with N pending "next tile" chunks
// guarantees the current tile's chunks completed (pipelined double-buffering).
template <int N>
static __device__ __forceinline__ void wait_asynccnt() {
  if constexpr (N == 0) asm volatile("s_wait_asynccnt 0" ::: "memory");
  else if constexpr (N == 1) asm volatile("s_wait_asynccnt 1" ::: "memory");
  else asm volatile("s_wait_asynccnt 2" ::: "memory");
}

// ---------------- fp32 -> bf16 convert ----------------
__global__ void cvt_kernel(const float* __restrict__ in, bf16* __restrict__ out, int n) {
  int i = blockIdx.x * 256 + threadIdx.x;
  if (i < n) out[i] = (bf16)in[i];
}

// ---------------- LayerNorm (faithful: unbiased var, NO sqrt) ----------------
__global__ void __launch_bounds__(256)
ln_kernel(const float* __restrict__ x, const float* __restrict__ sc,
          const float* __restrict__ sh, bf16* __restrict__ out) {
  __shared__ float red[256];
  const int row = blockIdx.x;
  const int tid = threadIdx.x;
  const float* xr = x + (size_t)row * Dm;
  float vals[Dm / 256];
  float s = 0.f, q = 0.f;
#pragma unroll
  for (int i = 0; i < Dm / 256; ++i) {
    float v = xr[tid + i * 256];
    vals[i] = v; s += v; q += v * v;
  }
  red[tid] = s; __syncthreads();
  for (int off = 128; off > 0; off >>= 1) {
    if (tid < off) red[tid] += red[tid + off];
    __syncthreads();
  }
  const float sum = red[0]; __syncthreads();
  red[tid] = q; __syncthreads();
  for (int off = 128; off > 0; off >>= 1) {
    if (tid < off) red[tid] += red[tid + off];
    __syncthreads();
  }
  const float qsum = red[0];
  const float mean = sum / (float)Dm;
  const float var  = (qsum - (float)Dm * mean * mean) / (float)(Dm - 1);
  const float inv  = 1.0f / (var + EPSV);   // faithful: no sqrt
  bf16* orow = out + (size_t)row * Dm;
#pragma unroll
  for (int i = 0; i < Dm / 256; ++i) {
    int c = tid + i * 256;
    orow[c] = (bf16)((vals[i] - mean) * inv * sc[c] + sh[c]);
  }
}

// ---------------- double-buffered WMMA GEMM ----------------
// C[M,N] = epi(A[M,K] @ W[N,K]^T + bias); block tile 64x256, wave tile 32x64, BK=32.
// A tiles: async global->LDS (b128/lane). W tiles: global->VGPR preload (latency
// overlapped with compute), then transposed scatter into LDS [k][n].
// EPI: 0 = bf16 out, 1 = exact GELU + bf16 out, 2 = +resid, fp32 out
template <int EPI>
__global__ void __launch_bounds__(256)
gemm_bf16(const bf16* __restrict__ A, const bf16* __restrict__ W,
          const float* __restrict__ bias, const float* __restrict__ resid,
          void* __restrict__ outp, int Mn, int Nn, int Kn) {
  (void)Mn;
  __shared__ __align__(16) bf16 As[2][64 * 32];    // [m][k]
  __shared__ __align__(16) bf16 Ws[2][32 * 264];   // [k][n], padded rows
  const int tid  = threadIdx.x;
  const int lane = tid & 31;
  const int wave = tid >> 5;
  const int wm = wave & 1, wn = wave >> 1;         // 2 x 4 wave grid
  const int m0 = blockIdx.y * 64;
  const int n0 = blockIdx.x * 256;

  const int ar = tid >> 2, ac = (tid & 3) * 8;     // A-stage: 16B per thread

  union WReg { uint4 q[4]; bf16 h[32]; };

  auto stageA = [&](int buf, int k0) {
    async_copy_b128(&As[buf][ar * 32 + ac], A + (size_t)(m0 + ar) * Kn + k0 + ac);
  };
  auto loadW = [&](WReg& wr, int k0) {
    const bf16* src = W + (size_t)(n0 + tid) * Kn + k0;
#pragma unroll
    for (int i = 0; i < 4; ++i) wr.q[i] = *(const uint4*)(src + i * 8);
  };
  auto storeW = [&](int buf, const WReg& wr) {
#pragma unroll
    for (int c = 0; c < 32; ++c) Ws[buf][c * 264 + tid] = wr.h[c];
  };

  const v8f vzero = {0.f, 0.f, 0.f, 0.f, 0.f, 0.f, 0.f, 0.f};
  v8f acc[2][4];
#pragma unroll
  for (int i = 0; i < 2; ++i)
#pragma unroll
    for (int j = 0; j < 4; ++j) acc[i][j] = vzero;

  auto compute = [&](int buf) {
    const bf16* Asb = &As[buf][0];
    const bf16* Wsb = &Ws[buf][0];
    v16bf a0 = lds_Afrag(Asb, wm * 32 + 0,  32, 0, lane);
    v16bf a1 = lds_Afrag(Asb, wm * 32 + 16, 32, 0, lane);
    v16bf bfr[4];
#pragma unroll
    for (int ni = 0; ni < 4; ++ni)
      bfr[ni] = lds_Bfrag(Wsb, 264, wn * 64 + ni * 16, lane);
#pragma unroll
    for (int ni = 0; ni < 4; ++ni) {
      acc[0][ni] = wmma_bf16(a0, bfr[ni], acc[0][ni]);
      acc[1][ni] = wmma_bf16(a1, bfr[ni], acc[1][ni]);
    }
  };

  const int NIT = Kn / 32;
  WReg wreg;
  stageA(0, 0);
  loadW(wreg, 0);
  int it = 0;
  for (; it < NIT - 1; ++it) {
    const int cur = it & 1;
    storeW(cur, wreg);
    stageA(cur ^ 1, (it + 1) * 32);     // async: streams behind compute
    WReg wnext;
    loadW(wnext, (it + 1) * 32);        // plain loads: retire during compute
    wait_asynccnt<1>();                 // current A tile done (in-order)
    __syncthreads();
    compute(cur);
    wreg = wnext;
    __syncthreads();
  }
  {
    const int cur = it & 1;
    storeW(cur, wreg);
    wait_asynccnt<0>();
    __syncthreads();
    compute(cur);
  }

  const int ml = lane & 15;
  const int halfo = (lane >> 4) << 3;
#pragma unroll
  for (int mi = 0; mi < 2; ++mi)
#pragma unroll
    for (int ni = 0; ni < 4; ++ni)
#pragma unroll
      for (int r = 0; r < 8; ++r) {
        int gm = m0 + wm * 32 + mi * 16 + r + halfo;
        int gn = n0 + wn * 64 + ni * 16 + ml;
        float v = acc[mi][ni][r];
        if (bias) v += bias[gn];
        if (EPI == 1) v = 0.5f * v * (1.0f + erff(v * 0.70710678118654752f));
        size_t o = (size_t)gm * Nn + gn;
        if (EPI == 2) ((float*)outp)[o] = v + resid[o];
        else          ((bf16*)outp)[o]  = (bf16)v;
      }
}

// ---------------- flash attention, transposed + double-buffered ----------------
// S^T = K_tile(32x64) . Q^T(64x16) ; O^T = V^T(64x32) . P^T(32x16)
// K tiles: async global->LDS natural layout. V tiles: VGPR preload + transposed
// scatter. Per-lane softmax stats (q = lane%16), one cross-half shuffle each.
__global__ void __launch_bounds__(128)
attn_kernel(const bf16* __restrict__ qg, const bf16* __restrict__ kg,
            const bf16* __restrict__ vg, bf16* __restrict__ ctx) {
  __shared__ __align__(16) bf16 Qt[64 * 72];       // [feat][q]
  __shared__ __align__(16) bf16 Ks[2][32 * 72];    // [kv][feat] natural (async dest)
  __shared__ __align__(16) bf16 Vt[2][64 * 36];    // [feat][kv]
  __shared__ __align__(16) bf16 Ps[4][32 * 18];    // per-wave P^T [kv][q]
  const int tid  = threadIdx.x;
  const int lane = tid & 31;
  const int wv   = tid >> 5;
  const int qt = blockIdx.x, h = blockIdx.y, b = blockIdx.z;
  const int tq0 = qt * 64;
  const int ml = lane & 15;
  const int halfo = (lane >> 4) << 3;

  // stage Q^T once for the block
  {
    const bf16* qbase = qg + ((size_t)(b * Tm + tq0)) * Dm + h * HDm;
    for (int idx = tid; idx < 64 * 64; idx += 128) {
      int j = idx >> 6, f = idx & 63;
      Qt[f * 72 + j] = qbase[(size_t)j * Dm + f];
    }
  }
  __syncthreads();
  v16bf qb[2];
#pragma unroll
  for (int kc = 0; kc < 2; ++kc)
    qb[kc] = lds_Bfrag(Qt + kc * 32 * 72, 72, wv * 16, lane);

  const bf16* kg0 = kg + ((size_t)(b * Tm)) * Dm + h * HDm;
  const bf16* vg0 = vg + ((size_t)(b * Tm)) * Dm + h * HDm;
  const int vf = tid & 63, vj0 = tid >> 6;     // V-stage mapping: fixed feat column

  auto stageK = [&](int buf, int kt) {
    const bf16* kbase = kg0 + (size_t)kt * 32 * Dm;
#pragma unroll
    for (int c = 0; c < 2; ++c) {
      int chunk = tid * 2 + c;                 // 0..255 : 16B chunks
      int j = chunk >> 3, cc = chunk & 7;
      async_copy_b128(&Ks[buf][j * 72 + cc * 8], kbase + (size_t)j * Dm + cc * 8);
    }
  };
  auto loadV = [&](bf16* vr, int kt) {
    const bf16* vbase = vg0 + (size_t)kt * 32 * Dm;
#pragma unroll
    for (int i = 0; i < 16; ++i)
      vr[i] = vbase[(size_t)(vj0 + 2 * i) * Dm + vf];
  };
  auto storeV = [&](int buf, const bf16* vr) {
#pragma unroll
    for (int i = 0; i < 16; ++i)
      Vt[buf][vf * 36 + vj0 + 2 * i] = vr[i];
  };

  const v8f vzero = {0.f, 0.f, 0.f, 0.f, 0.f, 0.f, 0.f, 0.f};
  v8f o[4];
#pragma unroll
  for (int i = 0; i < 4; ++i) o[i] = vzero;
  float mi_ = -INFINITY, li_ = 0.f;            // per-lane stats for q = ml
  bf16* Pl = &Ps[wv][0];

  auto compute = [&](int buf) {
    // S^T = K . Q^T
    v8f s0 = vzero, s1 = vzero;
#pragma unroll
    for (int kc = 0; kc < 2; ++kc) {
      v16bf ka0 = lds_Afrag(&Ks[buf][0], 0,  72, kc * 32, lane);
      v16bf ka1 = lds_Afrag(&Ks[buf][0], 16, 72, kc * 32, lane);
      s0 = wmma_bf16(ka0, qb[kc], s0);
      s1 = wmma_bf16(ka1, qb[kc], s1);
    }
    // online softmax over kv
    float tmax = -INFINITY;
#pragma unroll
    for (int r = 0; r < 8; ++r) tmax = fmaxf(tmax, fmaxf(s0[r], s1[r]));
    tmax *= 0.125f;                            // 1/sqrt(64)
    tmax = fmaxf(tmax, __shfl_xor(tmax, 16, 32));
    const float mn = fmaxf(mi_, tmax);
    const float al = __expf(mi_ - mn);
    float rs = 0.f;
#pragma unroll
    for (int r = 0; r < 8; ++r) {
      float p0 = __expf(s0[r] * 0.125f - mn);
      float p1 = __expf(s1[r] * 0.125f - mn);
      rs += p0 + p1;
      Pl[(r + halfo) * 18 + ml]      = (bf16)p0;
      Pl[(16 + r + halfo) * 18 + ml] = (bf16)p1;
    }
    rs += __shfl_xor(rs, 16, 32);
    li_ = li_ * al + rs;
    mi_ = mn;
#pragma unroll
    for (int i = 0; i < 4; ++i)
#pragma unroll
      for (int r = 0; r < 8; ++r) o[i][r] *= al;
    // O^T += V^T . P^T
    v16bf pb = lds_Bfrag(Pl, 18, 0, lane);
#pragma unroll
    for (int mc = 0; mc < 4; ++mc) {
      v16bf va = lds_Afrag(&Vt[buf][0], mc * 16, 36, 0, lane);
      o[mc] = wmma_bf16(va, pb, o[mc]);
    }
  };

  const int NT = Tm / 32;
  bf16 vreg[16];
  stageK(0, 0);
  loadV(vreg, 0);
  int it = 0;
  for (; it < NT - 1; ++it) {
    const int cur = it & 1;
    storeV(cur, vreg);
    stageK(cur ^ 1, it + 1);                   // async: streams behind compute
    bf16 vnext[16];
    loadV(vnext, it + 1);                      // plain loads: retire during compute
    wait_asynccnt<2>();                        // current K tile done (in-order)
    __syncthreads();
    compute(cur);
#pragma unroll
    for (int i = 0; i < 16; ++i) vreg[i] = vnext[i];
    __syncthreads();
  }
  {
    const int cur = it & 1;
    storeV(cur, vreg);
    wait_asynccnt<0>();
    __syncthreads();
    compute(cur);
  }

  // normalize + faithful transpose(0,1,3,2).reshape(B,T,D) scatter
  const float inv = 1.0f / li_;
#pragma unroll
  for (int mc = 0; mc < 4; ++mc)
#pragma unroll
    for (int r = 0; r < 8; ++r) {
      float val = o[mc][r] * inv;
      int d_ = mc * 16 + r + halfo;
      int tg = tq0 + wv * 16 + ml;
      size_t flat = (size_t)h * (HDm * Tm) + (size_t)d_ * Tm + tg;
      ctx[(size_t)b * Tm * Dm + flat] = (bf16)val;
    }
}

// ---------------- launcher ----------------
extern "C" void kernel_launch(void* const* d_in, const int* in_sizes, int n_in,
                              void* d_out, int out_size, void* d_ws, size_t ws_size,
                              hipStream_t stream) {
  (void)in_sizes; (void)n_in; (void)out_size; (void)ws_size;
  const float* x      = (const float*)d_in[0];
  const float* wq     = (const float*)d_in[1];
  const float* wk     = (const float*)d_in[2];
  const float* wv     = (const float*)d_in[3];
  const float* wo     = (const float*)d_in[4];
  const float* bo     = (const float*)d_in[5];
  const float* w_up   = (const float*)d_in[6];
  const float* b_up   = (const float*)d_in[7];
  const float* w_down = (const float*)d_in[8];
  const float* b_down = (const float*)d_in[9];
  const float* scale1 = (const float*)d_in[10];
  const float* shift1 = (const float*)d_in[11];
  const float* scale2 = (const float*)d_in[12];
  const float* shift2 = (const float*)d_in[13];

  char* ws = (char*)d_ws;
  size_t off = 0;
  auto alloc = [&](size_t bytes) -> char* {
    char* p = ws + off;
    off += (bytes + 255) & ~(size_t)255;
    return p;
  };
  bf16* wq_b  = (bf16*)alloc((size_t)Dm * Dm * 2);
  bf16* wk_b  = (bf16*)alloc((size_t)Dm * Dm * 2);
  bf16* wv_b  = (bf16*)alloc((size_t)Dm * Dm * 2);
  bf16* wo_b  = (bf16*)alloc((size_t)Dm * Dm * 2);
  bf16* wup_b = (bf16*)alloc((size_t)Fm * Dm * 2);
  bf16* wdn_b = (bf16*)alloc((size_t)Dm * Fm * 2);
  bf16* h_b   = (bf16*)alloc((size_t)Mm * Dm * 2);
  bf16* q_b   = (bf16*)alloc((size_t)Mm * Dm * 2);
  bf16* k_b   = (bf16*)alloc((size_t)Mm * Dm * 2);
  bf16* v_b   = (bf16*)alloc((size_t)Mm * Dm * 2);
  bf16* ctx_b = (bf16*)alloc((size_t)Mm * Dm * 2);
  float* x1   = (float*)alloc((size_t)Mm * Dm * 4);
  bf16* u_b   = (bf16*)alloc((size_t)Mm * Fm * 2);

  // 1) weights -> bf16
  cvt_kernel<<<(Dm * Dm) / 256, 256, 0, stream>>>(wq, wq_b, Dm * Dm);
  cvt_kernel<<<(Dm * Dm) / 256, 256, 0, stream>>>(wk, wk_b, Dm * Dm);
  cvt_kernel<<<(Dm * Dm) / 256, 256, 0, stream>>>(wv, wv_b, Dm * Dm);
  cvt_kernel<<<(Dm * Dm) / 256, 256, 0, stream>>>(wo, wo_b, Dm * Dm);
  cvt_kernel<<<(Fm * Dm) / 256, 256, 0, stream>>>(w_up, wup_b, Fm * Dm);
  cvt_kernel<<<(Dm * Fm) / 256, 256, 0, stream>>>(w_down, wdn_b, Dm * Fm);

  // 2) LN1 -> h_b
  ln_kernel<<<Mm, 256, 0, stream>>>(x, scale1, shift1, h_b);

  // 3) Q,K,V projections (no bias)
  dim3 gDD(Dm / 256, Mm / 64);
  gemm_bf16<0><<<gDD, 256, 0, stream>>>(h_b, wq_b, nullptr, nullptr, q_b, Mm, Dm, Dm);
  gemm_bf16<0><<<gDD, 256, 0, stream>>>(h_b, wk_b, nullptr, nullptr, k_b, Mm, Dm, Dm);
  gemm_bf16<0><<<gDD, 256, 0, stream>>>(h_b, wv_b, nullptr, nullptr, v_b, Mm, Dm, Dm);

  // 4) flash attention -> ctx_b (scrambled layout, faithful)
  attn_kernel<<<dim3(Tm / 64, Hm, Bm), 128, 0, stream>>>(q_b, k_b, v_b, ctx_b);

  // 5) O-proj + bias + residual(x) -> x1 (fp32)
  gemm_bf16<2><<<gDD, 256, 0, stream>>>(ctx_b, wo_b, bo, x, x1, Mm, Dm, Dm);

  // 6) LN2 -> h_b
  ln_kernel<<<Mm, 256, 0, stream>>>(x1, scale2, shift2, h_b);

  // 7) FFN up + GELU -> u_b
  gemm_bf16<1><<<dim3(Fm / 256, Mm / 64), 256, 0, stream>>>(h_b, wup_b, b_up, nullptr,
                                                            u_b, Mm, Fm, Dm);
  // 8) FFN down + bias + residual(x1) -> d_out (fp32)
  gemm_bf16<2><<<gDD, 256, 0, stream>>>(u_b, wdn_b, b_down, x1, (float*)d_out, Mm, Dm, Fm);
}
